// SpGraphAttentionLayer_rel_31430570672196
// MI455X (gfx1250) — matrix operand order repaired and checked by reading, not compiled, not run on hardware
//
#include <hip/hip_runtime.h>

typedef __attribute__((ext_vector_type(16))) __bf16 v16bf;
typedef __attribute__((ext_vector_type(8)))  float  v8f;

#define GAT_ALPHA 0.2f
#define F    128        // IN_F == OUT_F
#define KDIM 256        // IN_F + OUT_F
#define EPB  16         // edges per block (one M-tile)
#define SHA_STRIDE (KDIM + 8)   // ushort stride, keeps 16B alignment, rotates banks
#define SHM_STRIDE (F + 4)      // float stride, rotates banks

__device__ __forceinline__ unsigned short f32_to_bf16_rne(float f) {
    unsigned int u = __float_as_uint(f);
    unsigned int r = u + 0x7FFFu + ((u >> 16) & 1u);
    return (unsigned short)(r >> 16);
}

// ---------------------------------------------------------------------------
// Pack a[128,256] (f32, row-major: a[o][k]) into per-fragment bf16 B layout.
// Fragment f = wave*8 + kstep (wave = 16-col output tile, kstep = K/32 step).
// B (32x16, K x N) bf16 wave32 layout: lanes 0-15 hold K={0..7,16..23} for col N=lane,
// lanes 16-31 hold K={8..15,24..31} for col N=lane-16; element i => K=khalf*8+(i&7)+(i>>3)*16.
// B[kf][n] = a[n][kf].
// ---------------------------------------------------------------------------
__global__ void gat_pack_b_kernel(const float* __restrict__ a,
                                  unsigned short* __restrict__ bpack) {
    int f    = blockIdx.x;        // 0..63
    int lane = threadIdx.x;       // 0..31
    int wave = f >> 3;
    int k    = f & 7;
    int n    = wave * 16 + (lane & 15);
    int khalf = lane >> 4;
    unsigned short* dst = bpack + ((size_t)f * 32 + lane) * 16;
    const float*    src = a + (size_t)n * KDIM + k * 32 + khalf * 8;
#pragma unroll
    for (int i = 0; i < 16; ++i) dst[i] = f32_to_bf16_rne(src[(i & 7) + (i >> 3) * 16]);
}

__global__ void gat_zero_kernel(float* __restrict__ p, long long n) {
    long long i = (long long)blockIdx.x * blockDim.x + threadIdx.x;
    if (i < n) p[i] = 0.0f;
}

// ---------------------------------------------------------------------------
// Main fused edge kernel: 256 threads (8 waves) per 16-edge block.
// ---------------------------------------------------------------------------
__global__ void __launch_bounds__(256)
gat_edge_kernel(const float* __restrict__ x,
                const int*   __restrict__ edge_src,      // edge[0][:]
                const float* __restrict__ edge_embed,
                const unsigned short* __restrict__ bpack,
                const float* __restrict__ a2,
                float* __restrict__ hprime,              // d_out accumulator
                float* __restrict__ rowsum,
                int E) {
    __shared__ unsigned short shA[EPB][SHA_STRIDE];  // edge_h in bf16 (A-matrix)
    __shared__ float shM[EPB][SHM_STRIDE];           // edge_m f32
    __shared__ float shPow[EPB];
    __shared__ float shEE[EPB];
    __shared__ int   shSrc[EPB];
    __shared__ float shA2[F];

    const int tid  = threadIdx.x;
    const int lane = tid & 31;
    const int wave = tid >> 5;                 // 0..7 => output column tile
    const long long ebase = (long long)blockIdx.x * EPB;

    // ---- Phase 1: gather + convert to bf16 A-matrix in LDS -----------------
    const int row = tid >> 4;                  // 0..15 edge within block
    const int cb  = (tid & 15) * 8;            // 8-col chunk
    const long long e  = ebase + row;
    const long long ec = (e < E) ? e : (long long)(E - 1);
    const int s = edge_src[ec];

    if (tid < EPB) {
        shSrc[tid] = edge_src[(ebase + tid < E) ? (ebase + tid) : (long long)(E - 1)];
        shPow[tid] = 0.0f;
    }
    if (tid < F) shA2[tid] = a2[tid];

    {
        const float4* xp = (const float4*)(x + (size_t)s * F + cb);
        float4 v0 = xp[0], v1 = xp[1];
        const float4* ep = (const float4*)(edge_embed + (size_t)ec * F + cb);
        float4 w0 = ep[0], w1 = ep[1];
        unsigned short* da = &shA[row][cb];
        da[0] = f32_to_bf16_rne(v0.x); da[1] = f32_to_bf16_rne(v0.y);
        da[2] = f32_to_bf16_rne(v0.z); da[3] = f32_to_bf16_rne(v0.w);
        da[4] = f32_to_bf16_rne(v1.x); da[5] = f32_to_bf16_rne(v1.y);
        da[6] = f32_to_bf16_rne(v1.z); da[7] = f32_to_bf16_rne(v1.w);
        unsigned short* db = &shA[row][F + cb];
        db[0] = f32_to_bf16_rne(w0.x); db[1] = f32_to_bf16_rne(w0.y);
        db[2] = f32_to_bf16_rne(w0.z); db[3] = f32_to_bf16_rne(w0.w);
        db[4] = f32_to_bf16_rne(w1.x); db[5] = f32_to_bf16_rne(w1.y);
        db[6] = f32_to_bf16_rne(w1.z); db[7] = f32_to_bf16_rne(w1.w);
    }
    __syncthreads();

    // ---- Phase 2: edge_m tile = A(16x256 bf16) x B(256x16 bf16) ------------
    // A 16-bit layout (16x32 per step): lanes 0-15 => rows 0-15 holding
    // K = kb*32 + {0..7, 16..23}; lanes 16-31 => rows 0-15, K offset +8.
    v8f acc = {};
    {
        const int rowA = lane & 15;
        const int hi   = (lane >> 4) * 8;      // 0 or 8 (ushort offset)
        const unsigned short* arow = &shA[rowA][0];
#pragma unroll
        for (int k = 0; k < 8; ++k) {
            union { uint4 q[2]; v16bf v; } A, B;
            A.q[0] = *(const uint4*)(arow + k * 32 + hi);
            A.q[1] = *(const uint4*)(arow + k * 32 + 16 + hi);
            const uint4* bp =
                (const uint4*)(bpack + ((size_t)(wave * 8 + k) * 32 + lane) * 16);
            B.q[0] = bp[0];
            B.q[1] = bp[1];
            acc = __builtin_amdgcn_wmma_f32_16x16x32_bf16(
                false, A.v, false, B.v, (short)0, acc, false, false);
        }
    }

    // ---- Phase 3: spill accumulators to LDS --------------------------------
    // C/D layout: VGPR r, lanes 0-15 => M=r, N=lane; lanes 16-31 => M=r+8.
    {
        const int ncol = wave * 16 + (lane & 15);
        const int mb   = (lane < 16) ? 0 : 8;
#pragma unroll
        for (int r = 0; r < 8; ++r) shM[mb + r][ncol] = acc[r];
    }
    __syncthreads();

    // ---- Phase 4: logits: pow[e] = sum_o edge_m[e][o] * a2[o] --------------
    {
        const int le    = tid & 15;
        const int base  = (tid >> 4) * 8;
        float p = 0.0f;
#pragma unroll
        for (int j = 0; j < 8; ++j) p += shM[le][base + j] * shA2[base + j];
        atomicAdd(&shPow[le], p);              // LDS ds_add_f32
    }
    __syncthreads();

    // ---- Phase 5: edge_e = exp(-leaky_relu(pow)); rowsum scatter -----------
    if (tid < EPB) {
        float pw = shPow[tid];
        float lr = (pw > 0.0f) ? pw : GAT_ALPHA * pw;
        float ev = __expf(-lr);
        shEE[tid] = ev;
        if (ebase + tid < E) unsafeAtomicAdd(&rowsum[shSrc[tid]], ev);
    }
    __syncthreads();

    // ---- Phase 6: h_prime scatter: out[src[e]] += ee[e] * edge_m[e] --------
    if (e < E) {
        float ev = shEE[row];
        float* dst = hprime + (size_t)s * F + cb;
#pragma unroll
        for (int j = 0; j < 8; ++j)
            unsafeAtomicAdd(&dst[j], ev * shM[row][cb + j]);
    }
}

__global__ void gat_finalize_kernel(float* __restrict__ out,
                                    const float* __restrict__ rowsum,
                                    long long total) {
    long long i = (long long)blockIdx.x * blockDim.x + threadIdx.x;
    if (i < total) {
        float r = rowsum[i >> 7];
        r = (r == 0.0f) ? 1e-12f : r;
        out[i] /= r;
    }
}

// ---------------------------------------------------------------------------
extern "C" void kernel_launch(void* const* d_in, const int* in_sizes, int n_in,
                              void* d_out, int out_size, void* d_ws, size_t ws_size,
                              hipStream_t stream) {
    const float* x          = (const float*)d_in[0];
    const int*   edge       = (const int*)d_in[1];     // [2, E]; row 0 = src
    const float* edge_embed = (const float*)d_in[2];
    const float* a          = (const float*)d_in[3];   // [128, 256]
    const float* a2         = (const float*)d_in[4];   // [1, 128]
    float* out = (float*)d_out;

    const int N = in_sizes[0] / F;       // 50000
    const int E = in_sizes[2] / F;       // 1600000

    // workspace layout: rowsum [N floats] | bpack [64*32*16 bf16 = 64KB]
    float* rowsum = (float*)d_ws;
    size_t rs_bytes = ((size_t)N * sizeof(float) + 255) & ~(size_t)255;
    unsigned short* bpack = (unsigned short*)((char*)d_ws + rs_bytes);

    // 1) zero output accumulator + rowsum
    {
        long long tot = (long long)N * F;
        gat_zero_kernel<<<(unsigned)((tot + 255) / 256), 256, 0, stream>>>(out, tot);
        gat_zero_kernel<<<(unsigned)((N + 255) / 256), 256, 0, stream>>>(rowsum, N);
    }
    // 2) pack weight matrix into bf16 B fragments (64 fragments x 32 lanes)
    gat_pack_b_kernel<<<64, 32, 0, stream>>>(a, bpack);
    // 3) fused edge attention
    {
        unsigned nblk = (unsigned)((E + EPB - 1) / EPB);
        gat_edge_kernel<<<nblk, 256, 0, stream>>>(x, edge, edge_embed, bpack, a2,
                                                  out, rowsum, E);
    }
    // 4) normalize
    {
        long long tot = (long long)N * F;
        gat_finalize_kernel<<<(unsigned)((tot + 255) / 256), 256, 0, stream>>>(out, rowsum, tot);
    }
}
